// SpiderConv_67164698575433
// MI455X (gfx1250) — matrix operand order repaired and checked by reading, not compiled, not run on hardware
//
#include <hip/hip_runtime.h>

typedef __attribute__((ext_vector_type(16))) _Float16 v16h;
typedef __attribute__((ext_vector_type(2)))  _Float16 v2h;
typedef __attribute__((ext_vector_type(8)))  float    v8f;

#define B_    4
#define N_    8192
#define K_    20
#define C_    64
#define T_    5
#define O_    128
#define CT_   (C_ * T_)        // 320
#define J_    (K_ * CT_)       // 6400 reduction length
#define NSTEP_      (J_ / 32)  // 200 WMMA K-steps total
#define STEPS_PER_K (CT_ / 32) // 10 K-steps per k
#define XG_STRIDE   328        // halves per row (320 + 8 pad -> conflict-free b128)
#define MTILES      4          // M-tiles per block
#define BLK_PTS     (16 * MTILES)  // 64 points per block

// ---------------------------------------------------------------------------
// Prep: conv_w (O, CT, K) f32  ->  f16 B-matrix fragments in d_ws.
// Fragment layout: halfword index = ((step*8 + otile)*32 + lane)*16 + h
// B operand (32x16, 16-bit): lane l holds column n = l&15; kk = (l>>4)*8 + (h&7) + (h>>3)*16.
// ---------------------------------------------------------------------------
__global__ __launch_bounds__(256) void spider_prep_wt(const float* __restrict__ conv_w,
                                                      _Float16* __restrict__ wt) {
    int e = blockIdx.x * 256 + threadIdx.x;          // 0 .. 819199
    if (e >= NSTEP_ * 8 * 32 * 16) return;
    int h    = e & 15;
    int lane = (e >> 4) & 31;
    int ot   = (e >> 9) & 7;
    int step = e >> 12;
    int kk = ((lane >> 4) * 8) + (h & 7) + ((h >> 3) * 16);
    int j  = step * 32 + kk;
    int k  = j / CT_;
    int cc = j - k * CT_;
    int o  = ot * 16 + (lane & 15);
    wt[e] = (_Float16)conv_w[(o * CT_ + cc) * K_ + k];
}

// ---------------------------------------------------------------------------
// Main: block = 64 points, 8 waves. Tile grid = 4 M-tiles x 8 O-tiles; wave w
// owns the 2x2 quad: mt in {2*(w>>2), +1}, ot in {2*(w&3), +1}. Each A
// fragment feeds 2 WMMAs and each B fragment feeds 2 WMMAs (in-register
// reuse), balancing LDS vs L2 traffic at 1 ds_b128 + 1 global_b128 per WMMA.
// ---------------------------------------------------------------------------
__global__ __launch_bounds__(256) void spider_conv(
    const float* __restrict__ feat, const int* __restrict__ idx,
    const float* __restrict__ delta, const float* __restrict__ w_poly,
    const _Float16* __restrict__ wt, const float* __restrict__ conv_b,
    float* __restrict__ out)
{
    __shared__ float    gd[BLK_PTS * K_ * T_];           // 25.6 KB
    __shared__ _Float16 xg[2][BLK_PTS * XG_STRIDE];      // 2 x 42 KB

    const int tid  = threadIdx.x;
    const int lane = tid & 31;
    const int wave = tid >> 5;
    const int ptb  = blockIdx.x * BLK_PTS;

    const int mtb = (wave >> 2) * 2;                     // first M-tile of quad
    const int otb = (wave & 3) * 2;                      // first O-tile of quad

    // ---- Phase 1: g_d[m][k][t] = monomials(delta) @ w_poly ----
    for (int q = tid; q < BLK_PTS * K_; q += 256) {
        int m = q / K_, k = q - m * K_;
        int P = ptb + m;
        const float* d = delta + ((long)P * K_ + k) * 3;
        float X = d[0], Y = d[1], Z = d[2];
        float mono[20] = { X, Y, Z, X*Y*Z, X*Y, Y*Z, X*Z, 1.0f,
                           X*X, Y*Y, Z*Z, X*X*Y, X*Y*Y, X*X*Z, X*Z*Z,
                           Y*Y*Z, Y*Z*Z, X*X*X, Y*Y*Y, Z*Z*Z };
#pragma unroll
        for (int t = 0; t < T_; ++t) {
            float s = 0.0f;
#pragma unroll
            for (int p = 0; p < 20; ++p) s += mono[p] * w_poly[p * T_ + t];
            gd[(m * K_ + k) * T_ + t] = s;
        }
    }
    __syncthreads();

    // ---- xg builder: 512 tasks (m, 8-channel group), 2 per thread ----
    auto build_xg = [&](int k, _Float16* buf) {
#pragma unroll
        for (int rep = 0; rep < 2; ++rep) {
            int task = tid + rep * 256;
            int m  = task >> 3;
            int c8 = (task & 7) << 3;                 // 8 channels
            int P  = ptb + m;
            int jn = idx[P * K_ + k];
            int b  = P >> 13;                         // N = 8192
            const float* fr = feat + ((long)(b * N_ + jn)) * C_ + c8;
            float4 f0 = *(const float4*)fr;
            float4 f1 = *(const float4*)(fr + 4);
            const float* g = &gd[(m * K_ + k) * T_];
            float gt[5] = { g[0], g[1], g[2], g[3], g[4] };
            float fv[8] = { f0.x, f0.y, f0.z, f0.w, f1.x, f1.y, f1.z, f1.w };
            float p[40];
#pragma unroll
            for (int c = 0; c < 8; ++c)
#pragma unroll
                for (int t = 0; t < T_; ++t)
                    p[c * 5 + t] = fv[c] * gt[t];
            union { v2h h2[20]; uint4 q[5]; } tmp;
#pragma unroll
            for (int j = 0; j < 20; ++j) {
                auto r = __builtin_amdgcn_cvt_pkrtz(p[2 * j], p[2 * j + 1]);
                tmp.h2[j] = *(v2h*)&r;                // v_cvt_pk_rtz_f16_f32
            }
            uint4* dst = (uint4*)&buf[m * XG_STRIDE + c8 * 5];  // 16B aligned
#pragma unroll
            for (int q = 0; q < 5; ++q) dst[q] = tmp.q[q];
        }
    };

    v8f acc[2][2] = {};                               // [mi][oi] 16x16 tiles
    const int m_a = lane & 15;
    const int hi  = lane >> 4;

    build_xg(0, xg[0]);
    __syncthreads();

    for (int k = 0; k < K_; ++k) {
        const int cur = k & 1;
        if (k + 1 < K_) {
            build_xg(k + 1, xg[cur ^ 1]);
            // prefetch next k's B fragments into near caches
            const _Float16* nb =
                wt + ((size_t)((k + 1) * STEPS_PER_K * 8 + otb)) * 512 + lane * 16;
            __builtin_prefetch(nb, 0, 3);
            __builtin_prefetch(nb + 512, 0, 3);
        }

        const _Float16* wb0 =
            wt + ((size_t)(k * STEPS_PER_K * 8 + otb)) * 512 + lane * 16;
        const _Float16* abuf = xg[cur];

#pragma unroll
        for (int s = 0; s < STEPS_PER_K; ++s) {
            v16h b0 = *(const v16h*)(wb0 + (size_t)s * 8 * 512);
            v16h b1 = *(const v16h*)(wb0 + (size_t)s * 8 * 512 + 512);
#pragma unroll
            for (int mi = 0; mi < 2; ++mi) {
                v16h a;
                const _Float16* arow =
                    &abuf[((mtb + mi) * 16 + m_a) * XG_STRIDE + s * 32 + hi * 8];
                ((uint4*)&a)[0] = *(const uint4*)(arow);
                ((uint4*)&a)[1] = *(const uint4*)(arow + 16);
                acc[mi][0] = __builtin_amdgcn_wmma_f32_16x16x32_f16(
                                 false, a, false, b0, (short)0, acc[mi][0], false, false);
                acc[mi][1] = __builtin_amdgcn_wmma_f32_16x16x32_f16(
                                 false, a, false, b1, (short)0, acc[mi][1], false, false);
            }
        }
        __syncthreads();
    }

    // ---- epilogue: bias + ReLU, store per C/D VGPR layout ----
#pragma unroll
    for (int oi = 0; oi < 2; ++oi) {
        const int n_o = (otb + oi) * 16 + (lane & 15);
        const float bias = conv_b[n_o];
#pragma unroll
        for (int mi = 0; mi < 2; ++mi) {
#pragma unroll
            for (int r = 0; r < 8; ++r) {
                int m = (mtb + mi) * 16 + hi * 8 + r;
                float v = acc[mi][oi][r] + bias;
                out[(size_t)(ptb + m) * O_ + n_o] = v > 0.0f ? v : 0.0f;
            }
        }
    }
}

extern "C" void kernel_launch(void* const* d_in, const int* in_sizes, int n_in,
                              void* d_out, int out_size, void* d_ws, size_t ws_size,
                              hipStream_t stream) {
    const float* feat   = (const float*)d_in[0];
    const int*   idx    = (const int*)d_in[1];
    const float* delta  = (const float*)d_in[2];
    const float* w_poly = (const float*)d_in[3];
    const float* conv_w = (const float*)d_in[4];
    const float* conv_b = (const float*)d_in[5];
    float*    out = (float*)d_out;
    _Float16* wt  = (_Float16*)d_ws;              // 819200 * 2 B = 1.6 MB scratch

    (void)in_sizes; (void)n_in; (void)out_size; (void)ws_size;

    const int prep_elems = NSTEP_ * 8 * 32 * 16;  // 819200
    spider_prep_wt<<<(prep_elems + 255) / 256, 256, 0, stream>>>(conv_w, wt);
    spider_conv<<<(B_ * N_) / BLK_PTS, 256, 0, stream>>>(feat, idx, delta, w_poly,
                                                         wt, conv_b, out);
}
